// _TaskConditionedPointToSupernodeMessagePassing_65781719106308
// MI455X (gfx1250) — matrix operand order repaired and checked by reading, not compiled, not run on hardware
//
#include <hip/hip_runtime.h>
#include <hip/hip_bf16.h>

typedef __attribute__((ext_vector_type(16))) _Float16 v16h;
typedef __attribute__((ext_vector_type(8)))  float    v8f;

union Frag16 { v16h h; unsigned int u[8]; };

#define B_   8
#define N_   100000
#define M_   2048
#define K_   32
#define D_   128
#define KP1  160          // GEMM1 K-dim padded 132 -> 160 (5 chunks of 32)
#define EPS_ 1e-5f
#define TM   8            // supernodes per workgroup

// ---- dynamic LDS layout (bytes) ----
#define OFF_WT1  0                            // [128][160] f16 (W1^T, k-contig)
#define OFF_WT2  (OFF_WT1 + 128*KP1*2)        // [128][128] f16 (W2^T)
#define OFF_X    (OFF_WT2 + 128*128*2)        // [32][160] f16 gathered inputs
#define OFF_H    (OFF_X   + 32*KP1*2)         // [32][128] f16 hidden
#define OFF_AGG  (OFF_H   + 32*128*2)         // [128] f32 column sums
#define OFF_FILM (OFF_AGG + 128*4)            // [256] f32 gamma|beta
#define OFF_B1   (OFF_FILM + 256*4)           // [128] f32
#define OFF_B2   (OFF_B1  + 128*4)            // [128] f32
#define OFF_MASK (OFF_B2  + 128*4)            // [32] f32
#define OFF_RED  (OFF_MASK + 32*4)            // [8] f32 reduction scratch
#define SMEM_BYTES (OFF_RED + 8*4)            // 94880 bytes

__global__ void __launch_bounds__(256)
supernode_mp_kernel(const float* __restrict__ point_feat,
                    const float* __restrict__ point_xyz,
                    const float* __restrict__ supernode_xyz,
                    const int*   __restrict__ neighbor_idx,
                    const int*   __restrict__ neighbor_mask,
                    const float* __restrict__ supernode_init,
                    const float* __restrict__ task_emb,
                    const float* __restrict__ W1, const float* __restrict__ b1,
                    const float* __restrict__ W2, const float* __restrict__ b2,
                    const float* __restrict__ Wf, const float* __restrict__ bf,
                    const float* __restrict__ ln_g, const float* __restrict__ ln_b,
                    float* __restrict__ out)
{
    extern __shared__ char smem[];
    _Float16* wT1   = (_Float16*)(smem + OFF_WT1);
    _Float16* wT2   = (_Float16*)(smem + OFF_WT2);
    _Float16* xs    = (_Float16*)(smem + OFF_X);
    _Float16* hs    = (_Float16*)(smem + OFF_H);
    float*    aggs  = (float*)(smem + OFF_AGG);
    float*    film  = (float*)(smem + OFF_FILM);
    float*    b1s   = (float*)(smem + OFF_B1);
    float*    b2s   = (float*)(smem + OFF_B2);
    float*    masks = (float*)(smem + OFF_MASK);
    float*    red   = (float*)(smem + OFF_RED);

    const int t    = threadIdx.x;
    const int lane = t & 31;
    const int wid  = t >> 5;
    const int wg   = blockIdx.x;
    const int b    = wg >> 8;              // 256 groups per batch (2048/TM)
    const int m0   = (wg & 255) * TM;

    // ---- stage weights (transposed, f16) + biases + FiLM vector ----
    for (int i = t; i < 128 * KP1; i += 256) {
        int n = i / KP1, k = i % KP1;
        float v = (k < 132) ? W1[k * 128 + n] : 0.f;
        wT1[i] = (_Float16)v;
    }
    for (int i = t; i < 128 * 128; i += 256) {
        int n = i >> 7, k = i & 127;
        wT2[i] = (_Float16)W2[k * 128 + n];
    }
    if (t < 128) { b1s[t] = b1[t]; b2s[t] = b2[t]; }
    {   // gamma|beta = task_emb[b] @ Wf + bf  (256 outputs, one per thread)
        float acc = bf[t];
        for (int i = 0; i < 128; ++i) acc += task_emb[b * 128 + i] * Wf[i * 256 + t];
        film[t] = acc;
    }
    // zero constant pad columns of x (132..159) once
    for (int i = t; i < 32 * (KP1 - 132); i += 256) {
        int r = i / (KP1 - 132), c = 132 + i % (KP1 - 132);
        xs[r * KP1 + c] = (_Float16)0.f;
    }
    __syncthreads();

    const int rt    = wid >> 2;            // row tile (0/1)
    const int ct0   = (wid & 3);           // column tiles
    const int ct1   = (wid & 3) + 4;
    const int lrow  = lane & 15;
    const int khalf = lane >> 4;

    for (int mi = 0; mi < TM; ++mi) {
        const int m = m0 + mi;

        // ---- gather: 8 threads per neighbor, 16 feats each ----
        if (t < 128) aggs[t] = 0.f;
        {
            const int k   = t >> 3, sub = t & 7;
            const int idx = neighbor_idx[((long long)(b * M_) + m) * K_ + k];
            const float* src = point_feat + ((long long)b * N_ + idx) * D_ + sub * 16;
            _Float16* dst = xs + k * KP1 + sub * 16;
            #pragma unroll
            for (int j = 0; j < 16; j += 4) {
                float4 f = *(const float4*)(src + j);
                dst[j]   = (_Float16)f.x; dst[j+1] = (_Float16)f.y;
                dst[j+2] = (_Float16)f.z; dst[j+3] = (_Float16)f.w;
            }
            if (sub == 0) {
                const float* pxyz = point_xyz + ((long long)b * N_ + idx) * 3;
                const float* sxyz = supernode_xyz + ((long long)(b * M_) + m) * 3;
                float rx = pxyz[0] - sxyz[0];
                float ry = pxyz[1] - sxyz[1];
                float rz = pxyz[2] - sxyz[2];
                float dd = sqrtf(rx*rx + ry*ry + rz*rz);
                xs[k*KP1 + 128] = (_Float16)rx;
                xs[k*KP1 + 129] = (_Float16)ry;
                xs[k*KP1 + 130] = (_Float16)rz;
                xs[k*KP1 + 131] = (_Float16)dd;
                masks[k] = neighbor_mask[((long long)(b * M_) + m) * K_ + k] ? 1.f : 0.f;
            }
        }
        __syncthreads();

        // ---- GEMM1: [32,160]x[160,128] via v_wmma_f32_16x16x32_f16 ----
        v8f acc0 = {}, acc1 = {};
        for (int kk = 0; kk < 5; ++kk) {
            const int kb = kk * 32;
            Frag16 a, bA, bB;
            const _Float16* arow = xs + (rt * 16 + lrow) * KP1 + kb;
            #pragma unroll
            for (int p = 0; p < 8; ++p) {
                int ka = khalf * 16 + 2 * p;
                a.u[p] = *(const unsigned int*)(arow + ka);
            }
            const _Float16* br0 = wT1 + (ct0 * 16 + lrow) * KP1 + kb + khalf * 16;
            const _Float16* br1 = wT1 + (ct1 * 16 + lrow) * KP1 + kb + khalf * 16;
            #pragma unroll
            for (int p = 0; p < 8; ++p) {
                bA.u[p] = *(const unsigned int*)(br0 + 2 * p);
                bB.u[p] = *(const unsigned int*)(br1 + 2 * p);
            }
            acc0 = __builtin_amdgcn_wmma_f32_16x16x32_f16(false, a.h, false, bA.h, (short)0, acc0, false, false);
            acc1 = __builtin_amdgcn_wmma_f32_16x16x32_f16(false, a.h, false, bB.h, (short)0, acc1, false, false);
        }
        // SiLU(b1 + .) -> f16 hidden in LDS
        #pragma unroll
        for (int r = 0; r < 8; ++r) {
            int mr = rt * 16 + khalf * 8 + r;
            int n0 = ct0 * 16 + lrow, n1 = ct1 * 16 + lrow;
            float x0 = acc0[r] + b1s[n0];
            float x1 = acc1[r] + b1s[n1];
            hs[mr * 128 + n0] = (_Float16)(x0 / (1.f + __expf(-x0)));
            hs[mr * 128 + n1] = (_Float16)(x1 / (1.f + __expf(-x1)));
        }
        __syncthreads();

        // ---- GEMM2: [32,128]x[128,128] ----
        v8f c0 = {}, c1 = {};
        for (int kk = 0; kk < 4; ++kk) {
            const int kb = kk * 32;
            Frag16 a, bA, bB;
            const _Float16* arow = hs + (rt * 16 + lrow) * 128 + kb;
            #pragma unroll
            for (int p = 0; p < 8; ++p) {
                int ka = khalf * 16 + 2 * p;
                a.u[p] = *(const unsigned int*)(arow + ka);
            }
            const _Float16* br0 = wT2 + (ct0 * 16 + lrow) * 128 + kb + khalf * 16;
            const _Float16* br1 = wT2 + (ct1 * 16 + lrow) * 128 + kb + khalf * 16;
            #pragma unroll
            for (int p = 0; p < 8; ++p) {
                bA.u[p] = *(const unsigned int*)(br0 + 2 * p);
                bB.u[p] = *(const unsigned int*)(br1 + 2 * p);
            }
            c0 = __builtin_amdgcn_wmma_f32_16x16x32_f16(false, a.h, false, bA.h, (short)0, c0, false, false);
            c1 = __builtin_amdgcn_wmma_f32_16x16x32_f16(false, a.h, false, bB.h, (short)0, c1, false, false);
        }

        // ---- masked column sums (over K=32 neighbor rows) ----
        {
            float p0 = 0.f, p1 = 0.f;
            int n0 = ct0 * 16 + lrow, n1 = ct1 * 16 + lrow;
            #pragma unroll
            for (int r = 0; r < 8; ++r) {
                int mr = rt * 16 + khalf * 8 + r;
                float mk = masks[mr];
                p0 += (c0[r] + b2s[n0]) * mk;
                p1 += (c1[r] + b2s[n1]) * mk;
            }
            atomicAdd(&aggs[n0], p0);
            atomicAdd(&aggs[n1], p1);
        }
        __syncthreads();

        // ---- FiLM + residual + LayerNorm ----
        float hval = 0.f;
        if (t < 128) {
            float cnt = 0.f;
            for (int k = 0; k < 32; ++k) cnt += masks[k];
            float agg = aggs[t] / fmaxf(cnt, 1.f);
            agg = agg * (1.f + film[t]) + film[128 + t];
            hval = supernode_init[((long long)(b * M_) + m) * D_ + t] + agg;
        }
        float s = hval, s2 = hval * hval;
        #pragma unroll
        for (int off = 16; off >= 1; off >>= 1) {
            s  += __shfl_xor(s,  off, 32);
            s2 += __shfl_xor(s2, off, 32);
        }
        if (lane == 0 && wid < 4) { red[wid * 2] = s; red[wid * 2 + 1] = s2; }
        __syncthreads();
        if (t < 128) {
            float S  = red[0] + red[2] + red[4] + red[6];
            float S2 = red[1] + red[3] + red[5] + red[7];
            float mu  = S  * (1.f / 128.f);
            float var = S2 * (1.f / 128.f) - mu * mu;
            out[((long long)(b * M_) + m) * D_ + t] =
                (hval - mu) * rsqrtf(var + EPS_) * ln_g[t] + ln_b[t];
        }
        __syncthreads();
    }
}

extern "C" void kernel_launch(void* const* d_in, const int* in_sizes, int n_in,
                              void* d_out, int out_size, void* d_ws, size_t ws_size,
                              hipStream_t stream) {
    const float* point_feat     = (const float*)d_in[0];
    const float* point_xyz      = (const float*)d_in[1];
    const float* supernode_xyz  = (const float*)d_in[2];
    const int*   neighbor_idx   = (const int*)d_in[3];
    const int*   neighbor_mask  = (const int*)d_in[4];
    const float* supernode_init = (const float*)d_in[5];
    const float* task_emb       = (const float*)d_in[6];
    const float* W1 = (const float*)d_in[7];
    const float* b1 = (const float*)d_in[8];
    const float* W2 = (const float*)d_in[9];
    const float* b2 = (const float*)d_in[10];
    const float* Wf = (const float*)d_in[11];
    const float* bf = (const float*)d_in[12];
    const float* ln_g = (const float*)d_in[13];
    const float* ln_b = (const float*)d_in[14];
    float* outp = (float*)d_out;

    dim3 grid(B_ * M_ / TM);   // 2048 workgroups
    dim3 block(256);
    supernode_mp_kernel<<<grid, block, SMEM_BYTES, stream>>>(
        point_feat, point_xyz, supernode_xyz, neighbor_idx, neighbor_mask,
        supernode_init, task_emb, W1, b1, W2, b2, Wf, bf, ln_g, ln_b, outp);
}
